// MoELoRALayer_17746804867278
// MI455X (gfx1250) — matrix operand order repaired and checked by reading, not compiled, not run on hardware
//
#include <hip/hip_runtime.h>
#include <hip/hip_bf16.h>

typedef __bf16 bf16_t;
typedef __attribute__((ext_vector_type(16))) __bf16 v16bf;
typedef __attribute__((ext_vector_type(8)))  __bf16 v8bf;
typedef __attribute__((ext_vector_type(8)))  float  v8f;
typedef __attribute__((ext_vector_type(4)))  float  v4f;

#define B_    16
#define N_    577
#define D_    1024
#define E_    8
#define R_    16
#define C3_   3072
#define TOPK_ 2
#define NT_   37   // ceil(577/16)

// ---------------------------------------------------------------------------
// Kernel 1: router. One block per batch element. Exact f32 math so the top-2
// selection and softmax weights match the reference.
// ---------------------------------------------------------------------------
__global__ __launch_bounds__(256) void router_kernel(
    const float* __restrict__ tokens, const float* __restrict__ gate_w,
    const float* __restrict__ scaling, float* __restrict__ out,
    int* __restrict__ sel_e, float* __restrict__ sel_w)
{
    __shared__ float pooled[D_];
    __shared__ float red[256];
    __shared__ float logits[E_];
    const int b = blockIdx.x;
    const int t = threadIdx.x;
    const float* tb = tokens + (size_t)b * N_ * D_;

    // mean-pool over N=577 tokens
    for (int d = t; d < D_; d += 256) {
        float s = 0.f;
        for (int n = 0; n < N_; ++n) s += tb[(size_t)n * D_ + d];
        pooled[d] = s * (1.0f / (float)N_);
    }
    __syncthreads();

    // router logits: pooled @ gate_w  [D x E]
    float part[E_];
    #pragma unroll
    for (int e = 0; e < E_; ++e) part[e] = 0.f;
    for (int d = t; d < D_; d += 256) {
        const float p = pooled[d];
        #pragma unroll
        for (int e = 0; e < E_; ++e) part[e] += p * gate_w[(size_t)d * E_ + e];
    }
    for (int e = 0; e < E_; ++e) {
        red[t] = part[e];
        __syncthreads();
        for (int s = 128; s > 0; s >>= 1) {
            if (t < s) red[t] += red[t + s];
            __syncthreads();
        }
        if (t == 0) logits[e] = red[0];
        __syncthreads();
    }

    if (t == 0) {
        const size_t O_upd = (size_t)B_ * N_ * C3_;
        float* rl = out + O_upd;           // router_logits [B,E]
        float* ti = rl + B_ * E_;          // top_idx       [B,TOPK]
        float* ew = ti + B_ * TOPK_;       // expert_weights[B,E]

        int i0 = 0;
        for (int e = 1; e < E_; ++e) if (logits[e] > logits[i0]) i0 = e;
        int i1 = (i0 == 0) ? 1 : 0;
        for (int e = 0; e < E_; ++e)
            if (e != i0 && logits[e] > logits[i1]) i1 = e;

        const float v0 = logits[i0], v1 = logits[i1];
        const float e0 = __expf(v0 - v0), e1 = __expf(v1 - v0);
        const float inv = 1.f / (e0 + e1);
        const float w0 = e0 * inv, w1 = e1 * inv;

        for (int e = 0; e < E_; ++e) { rl[b * E_ + e] = logits[e]; ew[b * E_ + e] = 0.f; }
        ew[b * E_ + i0] = w0;
        ew[b * E_ + i1] = w1;
        ti[b * TOPK_ + 0] = (float)i0;
        ti[b * TOPK_ + 1] = (float)i1;

        sel_e[b * 2 + 0] = i0;
        sel_e[b * 2 + 1] = i1;
        sel_w[b * 2 + 0] = w0 * scaling[i0];   // fold softmax weight * lora scale
        sel_w[b * 2 + 1] = w1 * scaling[i1];
    }
}

// ---------------------------------------------------------------------------
// Kernel 2: importance / load reductions (tiny).
// ---------------------------------------------------------------------------
__global__ void tail_kernel(float* __restrict__ out)
{
    const size_t O_upd = (size_t)B_ * N_ * C3_;
    float* rl   = out + O_upd;
    float* ti   = rl + B_ * E_;
    float* ew   = ti + B_ * TOPK_;
    float* imp  = ew + B_ * E_;
    float* load = imp + E_;
    const int e = threadIdx.x;
    if (e < E_) {
        float s = 0.f, l = 0.f;
        for (int b = 0; b < B_; ++b) {
            s += ew[b * E_ + e];
            for (int k = 0; k < TOPK_; ++k)
                if ((int)ti[b * TOPK_ + k] == e) l += 1.f;
        }
        imp[e] = s;
        load[e] = l;
    }
}

// ---------------------------------------------------------------------------
// Kernel 3: down-projection. One wave per (b, expert-slot, 16-row n-tile).
// U[16n x 16r] = (w*scale) * tokens_tile[16x1024] @ lora_a[e]^T
// via 32x v_wmma_f32_16x16x32_bf16 over K=1024. Fragment layouts (§7.12.2):
//   A elem idx=2v+h -> K = idx + 8*half (idx<8) / idx-8 + 8*half + 16 (idx>=8)
//   B elem idx=2v+h -> K = same mapping as A
// Both are two contiguous 8-float runs per lane -> pure b128 traffic.
// ---------------------------------------------------------------------------
__global__ __launch_bounds__(32) void lora_down_kernel(
    const float* __restrict__ tokens, const float* __restrict__ lora_a,
    const int* __restrict__ sel_e, const float* __restrict__ sel_w,
    bf16_t* __restrict__ Uws)
{
    const int nt = blockIdx.x, slot = blockIdx.y, b = blockIdx.z;
    const int lane = threadIdx.x;
    const int half = lane >> 4;
    const int l16  = lane & 15;
    const int   e       = sel_e[b * 2 + slot];
    const float wscaled = sel_w[b * 2 + slot];

    int row = nt * 16 + l16;
    if (row >= N_) row = N_ - 1;   // clamp; dup rows masked at final store
    const float* arow = tokens + ((size_t)b * N_ + row) * D_;
    const float* brow = lora_a + ((size_t)e * R_ + l16) * D_;

    v8f acc = {};
    for (int kk = 0; kk < D_ / 32; ++kk) {
        const int kbase = kk * 32;
        __builtin_prefetch(arow + kbase + 128, 0, 0);  // global_prefetch_b8

        const v4f a0 = *(const v4f*)(arow + kbase + 8 * half);
        const v4f a1 = *(const v4f*)(arow + kbase + 8 * half + 4);
        const v4f a2 = *(const v4f*)(arow + kbase + 8 * half + 16);
        const v4f a3 = *(const v4f*)(arow + kbase + 8 * half + 20);
        const v4f b0 = *(const v4f*)(brow + kbase + 8 * half);
        const v4f b1 = *(const v4f*)(brow + kbase + 8 * half + 4);
        const v4f b2 = *(const v4f*)(brow + kbase + 8 * half + 16);
        const v4f b3 = *(const v4f*)(brow + kbase + 8 * half + 20);

        v16bf av, bv;
        #pragma unroll
        for (int i = 0; i < 4; ++i) {
            av[i]      = (bf16_t)a0[i];
            av[4 + i]  = (bf16_t)a1[i];
            av[8 + i]  = (bf16_t)a2[i];
            av[12 + i] = (bf16_t)a3[i];
            bv[i]      = (bf16_t)b0[i];
            bv[4 + i]  = (bf16_t)b1[i];
            bv[8 + i]  = (bf16_t)b2[i];
            bv[12 + i] = (bf16_t)b3[i];
        }
        acc = __builtin_amdgcn_wmma_f32_16x16x32_bf16(
            false, av, false, bv, (short)0, acc, false, false);
    }

    // D frag: lane L, VGPR v -> U[m = v + 8*half][r = l16]; store pre-scaled bf16
    bf16_t* up = Uws + (((size_t)b * 2 + slot) * NT_ + nt) * 256;
    #pragma unroll
    for (int v = 0; v < 8; ++v) {
        const int m = v + 8 * half;
        up[m * 16 + l16] = (bf16_t)(acc[v] * wscaled);
    }
}

// ---------------------------------------------------------------------------
// Kernel 4: up-projection with the two experts fused into one K=32 WMMA:
//   out = [w0*U0 | w1*U1] @ [B0 ; B1]
// A frag: K = idx + 16*half -> lane half 0 from U0, half 1 from U1 -> two b128 bf16 loads,
//         reused across all 8 C-tiles of the wave.
// B frag: K = idx + 16*half -> lane half selects expert; 16 contiguous f32 of
//         lora_b[e_half][c0+l16][0..15] -> four b128 loads, no divergence.
// ---------------------------------------------------------------------------
__global__ __launch_bounds__(128) void lora_up_kernel(
    const bf16_t* __restrict__ Uws, const float* __restrict__ lora_b,
    const int* __restrict__ sel_e, float* __restrict__ out)
{
    const int tid  = threadIdx.x;
    const int wave = tid >> 5;
    const int lane = tid & 31;
    const int half = lane >> 4;
    const int l16  = lane & 15;
    const int nt = blockIdx.y, b = blockIdx.z;
    const int ct_base = (blockIdx.x * 4 + wave) * 8;   // 8 C-tiles per wave

    const int e_half = sel_e[b * 2 + half];            // expert for my K-half

    // A fragment: load once, reuse for all C-tiles
    const bf16_t* up0 = Uws + (((size_t)b * 2 + 0) * NT_ + nt) * 256;
    const bf16_t* up1 = Uws + (((size_t)b * 2 + 1) * NT_ + nt) * 256;
    const v8bf u0 = *(const v8bf*)((half ? up1 : up0) + l16 * 16);
    const v8bf u1 = *(const v8bf*)((half ? up1 : up0) + l16 * 16 + 8);
    v16bf av;
    #pragma unroll
    for (int i = 0; i < 8; ++i) { av[i] = u0[i]; av[8 + i] = u1[i]; }

    const int n0 = nt * 16;
    for (int t = 0; t < 8; ++t) {
        const int c0 = (ct_base + t) * 16;
        const float* brow = lora_b + ((size_t)e_half * C3_ + c0 + l16) * R_;
        const v4f b0 = *(const v4f*)(brow + 0);
        const v4f b1 = *(const v4f*)(brow + 4);
        const v4f b2 = *(const v4f*)(brow + 8);
        const v4f b3 = *(const v4f*)(brow + 12);
        v16bf bv;
        #pragma unroll
        for (int i = 0; i < 4; ++i) {
            bv[i]      = (bf16_t)b0[i];
            bv[4 + i]  = (bf16_t)b1[i];
            bv[8 + i]  = (bf16_t)b2[i];
            bv[12 + i] = (bf16_t)b3[i];
        }
        v8f acc = {};
        acc = __builtin_amdgcn_wmma_f32_16x16x32_bf16(
            false, av, false, bv, (short)0, acc, false, false);

        // D frag: lane L, VGPR v -> out[n0 + v + 8*half][c0 + l16]
        #pragma unroll
        for (int v = 0; v < 8; ++v) {
            const int row = n0 + v + 8 * half;
            if (row < N_)
                out[((size_t)b * N_ + row) * C3_ + c0 + l16] = acc[v];
        }
    }
}

// ---------------------------------------------------------------------------
extern "C" void kernel_launch(void* const* d_in, const int* in_sizes, int n_in,
                              void* d_out, int out_size, void* d_ws, size_t ws_size,
                              hipStream_t stream)
{
    const float* tokens  = (const float*)d_in[0];
    const float* gate_w  = (const float*)d_in[1];
    const float* lora_a  = (const float*)d_in[2];
    const float* lora_b  = (const float*)d_in[3];
    const float* scaling = (const float*)d_in[4];
    float* out = (float*)d_out;

    // workspace: [0,128): sel_e (32 ints); [128,256): sel_w (32 f32);
    // [256, 256+606208): U tiles as bf16 [B][2][NT][16][16] (pre-scaled)
    int*    sel_e = (int*)d_ws;
    float*  sel_w = (float*)((char*)d_ws + 128);
    bf16_t* Uws   = (bf16_t*)((char*)d_ws + 256);

    router_kernel<<<B_, 256, 0, stream>>>(tokens, gate_w, scaling, out, sel_e, sel_w);
    tail_kernel<<<1, 32, 0, stream>>>(out);

    dim3 gdown(NT_, 2, B_);
    lora_down_kernel<<<gdown, 32, 0, stream>>>(tokens, lora_a, sel_e, sel_w, Uws);

    dim3 gup(6, NT_, B_);   // 6 blocks * 4 waves * 8 tiles = 192 C-tiles
    lora_up_kernel<<<gup, 128, 0, stream>>>(Uws, lora_b, sel_e, out);
}
